// REVAttention_2602750181464
// MI455X (gfx1250) — matrix operand order repaired and checked by reading, not compiled, not run on hardware
//
#include <hip/hip_runtime.h>

#define BB 2
#define HH 16
#define LL 2048
#define DD 128
#define KT 64   // keys per streamed tile

typedef __attribute__((ext_vector_type(16))) __bf16 v16bf;
typedef __attribute__((ext_vector_type(8)))  __bf16 v8bf;
typedef __attribute__((ext_vector_type(8)))  float  v8f;

// Round-to-nearest-even f32->bf16 (used once, for Q)
static __device__ __forceinline__ __bf16 f2bf_rne(float f) {
    unsigned u = __builtin_bit_cast(unsigned, f);
    u += 0x7fffu + ((u >> 16) & 1u);
    unsigned short s = (unsigned short)(u >> 16);
    return __builtin_bit_cast(__bf16, s);
}

// Truncating f32->bf16 (hot path; 1 shift, error within bf16 quantization)
static __device__ __forceinline__ __bf16 f2bf_rtz(float f) {
    unsigned u = __builtin_bit_cast(unsigned, f);
    return __builtin_bit_cast(__bf16, (unsigned short)(u >> 16));
}

// Truncating pack of two f32 into {bf16(b),bf16(a)} — 3 ops, and LLVM's
// byte-perm combiner usually fuses this into a single v_perm_b32.
static __device__ __forceinline__ unsigned pk_hi(float a, float b) {
    unsigned ua = __builtin_bit_cast(unsigned, a);
    unsigned ub = __builtin_bit_cast(unsigned, b);
    return (ua >> 16) | (ub & 0xffff0000u);
}

// Single v_exp_f32 (base-2 exponential, the native AMDGCN transcendental)
#define EXP2(x) __builtin_amdgcn_exp2f(x)

// XOR-shuffle via ds_swizzle immediate pattern: lane' = lane ^ m (and_mask=0x1F)
#define SWZ_XOR(x, pat) \
    __builtin_bit_cast(float, __builtin_amdgcn_ds_swizzle(__builtin_bit_cast(int, x), (pat)))

// Scores are kept in the log2 domain: Q is pre-scaled by log2e/sqrt(D), the
// mask fill value is -10000*log2e, and all exponentials are single v_exp_f32
// (exp2). Softmax is shift/scale-invariant-equivalent, so results match the
// reference up to fp rounding.
#define MASK_NEG (-14426.950408889634f)   // -10000 * log2(e)

// One block = 8 wave32s = 128 query rows of one (b,h). Each wave owns a 16-row
// Q tile in registers, streams 64-key K/V tiles through LDS, flash-attention
// online softmax, bf16 WMMA math with f32 accumulation.
__global__ __launch_bounds__(256)
void revattn_kernel(const float* __restrict__ qg,
                    const float* __restrict__ kg,
                    const float* __restrict__ vg,
                    const int*   __restrict__ maskg,
                    float*       __restrict__ outg)
{
    // sK: K tile row-major [key][d], stride 136 (272B) -> conflict-free b128 gathers.
    // sVt: ReLU(V) transposed [d][phys_key], stride 72 (144B).
    // sP: per-wave P scratch [row][phys_key], stride 72.
    // Physical key relabel: phys = (key&15)*4 + (key>>4)  (applied to sVt & sP only;
    // softmax/PV are invariant under a consistent key permutation).
    __shared__ __align__(16) __bf16 sK [KT][136];
    __shared__ __align__(16) __bf16 sVt[DD][72];
    __shared__ __align__(16) __bf16 sP [8][16][72];

    const int tid  = threadIdx.x;
    const int wave = tid >> 5;
    const int lane = tid & 31;
    const int ln   = lane & 15;   // M / N index within 16
    const int hi   = lane >> 4;   // half-wave select

    const int bh   = blockIdx.x >> 4;    // 16 q-blocks per (b,h)
    const int qblk = blockIdx.x & 15;
    const int b    = bh >> 4;            // H == 16

    const size_t baseQKV = (size_t)bh * LL * DD;
    const int    q0      = qblk * 128 + wave * 16;
    const int* __restrict__ maskb = maskg + (size_t)b * LL * LL;

    // ---- Q tile (16x128) -> 4 WMMA-A operands, pre-scaled by log2e/sqrt(D).
    // A layout (16-bit 16x32): lane = row m=ln; elems 0..7 = K hi*8+0..7,
    // elems 8..15 = K 16+hi*8+0..7 within each 32-wide K chunk.
    const float scale = 0.088388347648318447f * 1.4426950408889634f;
    v16bf aQ[4];
    {
        const float* qrow = qg + baseQKV + (size_t)(q0 + ln) * DD;
        #pragma unroll
        for (int kc = 0; kc < 4; ++kc) {
            const float* p0 = qrow + kc * 32 + hi * 8;
            const float* p1 = qrow + kc * 32 + 16 + hi * 8;
            #pragma unroll
            for (int e = 0; e < 8; ++e) {
                aQ[kc][e]     = f2bf_rne(p0[e] * scale);
                aQ[kc][e + 8] = f2bf_rne(p1[e] * scale);
            }
        }
    }

    v8f o[8];                             // O accum: 8 chunks of 16 d-cols
    #pragma unroll
    for (int i = 0; i < 8; ++i)
        o[i] = (v8f){0.f,0.f,0.f,0.f,0.f,0.f,0.f,0.f};
    float mrow[8], srow[8];               // softmax stats (log2 domain), row r+8*hi
    #pragma unroll
    for (int r = 0; r < 8; ++r) { mrow[r] = -1e30f; srow[r] = 0.f; }

    const int ldky = tid >> 2;            // cooperative-load key   (0..63)
    const int ldd0 = (tid & 3) << 5;      // cooperative-load d0    (0,32,64,96)

    for (int kt = 0; kt < LL / KT; ++kt) {
        const int k0 = kt * KT;
        __syncthreads();
        // -- cooperative K tile (64 keys x 128 d), f32->bf16(RTZ), row-major
        {
            const float4* src =
                (const float4*)(kg + baseQKV + (size_t)(k0 + ldky) * DD + ldd0);
            unsigned* dst = (unsigned*)&sK[ldky][ldd0];
            #pragma unroll
            for (int j = 0; j < 8; ++j) {
                float4 f = src[j];
                dst[2 * j]     = pk_hi(f.x, f.y);
                dst[2 * j + 1] = pk_hi(f.z, f.w);
            }
        }
        // -- cooperative V tile, ReLU, transposed + key-permuted store
        {
            const int kphys = ((ldky & 15) << 2) | (ldky >> 4);
            const float4* src =
                (const float4*)(vg + baseQKV + (size_t)(k0 + ldky) * DD + ldd0);
            #pragma unroll
            for (int j = 0; j < 8; ++j) {
                float4 f = src[j];
                sVt[ldd0 + 4 * j + 0][kphys] = f2bf_rtz(fmaxf(f.x, 0.f));
                sVt[ldd0 + 4 * j + 1][kphys] = f2bf_rtz(fmaxf(f.y, 0.f));
                sVt[ldd0 + 4 * j + 2][kphys] = f2bf_rtz(fmaxf(f.z, 0.f));
                sVt[ldd0 + 4 * j + 3][kphys] = f2bf_rtz(fmaxf(f.w, 0.f));
            }
        }
        __syncthreads();

        // -- prefetch next tile's K/V slice (global_prefetch_b8) so HBM/L2
        // latency overlaps this tile's 32 WMMAs. Speculative-safe.
        if (kt + 1 < LL / KT) {
            const size_t noff = baseQKV + (size_t)(k0 + KT + ldky) * DD + ldd0;
            __builtin_prefetch(kg + noff, 0, 3);
            __builtin_prefetch(vg + noff, 0, 3);
        }

        // -- S = Q*K^T : 4 C tiles over 16-key groups, 16 bf16 WMMAs.
        // B layout (32x16): lane = key col; elems 0..7 = K rows hi*8 + 0..7,
        // elems 8..15 = K rows 16 + hi*8 + 0..7 -> two b128 LDS loads per operand.
        v8f c[4];
        #pragma unroll
        for (int t = 0; t < 4; ++t)
            c[t] = (v8f){0.f,0.f,0.f,0.f,0.f,0.f,0.f,0.f};
        #pragma unroll
        for (int kc = 0; kc < 4; ++kc) {
            #pragma unroll
            for (int t = 0; t < 4; ++t) {
                const __bf16* kr = &sK[t * 16 + ln][kc * 32 + hi * 8];
                v16bf bK;
                ((v8bf*)&bK)[0] = *(const v8bf*)(kr);
                ((v8bf*)&bK)[1] = *(const v8bf*)(kr + 16);
                c[t] = __builtin_amdgcn_wmma_f32_16x16x32_bf16(
                           false, aQ[kc], false, bK, (short)0, c[t], false, false);
            }
        }

        // -- mask + online softmax over 64 keys (log2 domain).
        // C layout: VGPR r = row r+8*hi, lane = key col within 16.
        #pragma unroll
        for (int r = 0; r < 8; ++r) {
            const int qrow = q0 + r + 8 * hi;
            const int* mp  = maskb + (size_t)qrow * LL + k0 + ln;
            float s0 = mp[0]  ? c[0][r] : MASK_NEG;
            float s1 = mp[16] ? c[1][r] : MASK_NEG;
            float s2 = mp[32] ? c[2][r] : MASK_NEG;
            float s3 = mp[48] ? c[3][r] : MASK_NEG;
            float mx = fmaxf(fmaxf(s0, s1), fmaxf(s2, s3));
            mx = fmaxf(mx, SWZ_XOR(mx, 0x041f));
            mx = fmaxf(mx, SWZ_XOR(mx, 0x081f));
            mx = fmaxf(mx, SWZ_XOR(mx, 0x101f));
            mx = fmaxf(mx, SWZ_XOR(mx, 0x201f));
            const float mnew  = fmaxf(mrow[r], mx);
            const float alpha = EXP2(mrow[r] - mnew);   // single v_exp_f32
            const float p0 = EXP2(s0 - mnew);
            const float p1 = EXP2(s1 - mnew);
            const float p2 = EXP2(s2 - mnew);
            const float p3 = EXP2(s3 - mnew);
            float ps = (p0 + p1) + (p2 + p3);
            ps += SWZ_XOR(ps, 0x041f);
            ps += SWZ_XOR(ps, 0x081f);
            ps += SWZ_XOR(ps, 0x101f);
            ps += SWZ_XOR(ps, 0x201f);
            srow[r] = srow[r] * alpha + ps;
            mrow[r] = mnew;
            #pragma unroll
            for (int ch = 0; ch < 8; ++ch) o[ch][r] *= alpha;  // rescale O
            // packed P spill: this lane's 4 keys land on phys cols ln*4..ln*4+3
            uint2 pk;
            pk.x = pk_hi(p0, p1);
            pk.y = pk_hi(p2, p3);
            *(uint2*)&sP[wave][r + 8 * hi][ln << 2] = pk;
        }

        // LDS in-order per wave; explicit wait + fence for cross-lane transpose.
        asm volatile("s_wait_dscnt 0x0" ::: "memory");

        // -- O += P*V : A = P(16x64) as two operands (phys key halves),
        // B = V chunk (32 phys keys x 16 d) from sVt. 16 bf16 WMMAs.
        v16bf aP[2];
        {
            const __bf16* prow = &sP[wave][ln][0];
            #pragma unroll
            for (int cc = 0; cc < 2; ++cc) {
                ((v8bf*)&aP[cc])[0] = *(const v8bf*)(prow + cc * 32 + hi * 8);
                ((v8bf*)&aP[cc])[1] = *(const v8bf*)(prow + cc * 32 + 16 + hi * 8);
            }
        }
        #pragma unroll
        for (int ch = 0; ch < 8; ++ch) {
            const __bf16* vrow = &sVt[ch * 16 + ln][0];
            #pragma unroll
            for (int cc = 0; cc < 2; ++cc) {
                v16bf bV;
                ((v8bf*)&bV)[0] = *(const v8bf*)(vrow + cc * 32 + hi * 8);
                ((v8bf*)&bV)[1] = *(const v8bf*)(vrow + cc * 32 + 16 + hi * 8);
                o[ch] = __builtin_amdgcn_wmma_f32_16x16x32_bf16(
                            false, aP[cc], false, bV, (short)0, o[ch], false, false);
            }
        }
    }

    // -- epilogue: divide by softmax denominator, store f32
    float* outb = outg + baseQKV;
    #pragma unroll
    for (int r = 0; r < 8; ++r) {
        const float inv = 1.0f / srow[r];
        float* orow = outb + (size_t)(q0 + r + 8 * hi) * DD;
        #pragma unroll
        for (int ch = 0; ch < 8; ++ch)
            orow[ch * 16 + ln] = o[ch][r] * inv;
    }
}

extern "C" void kernel_launch(void* const* d_in, const int* in_sizes, int n_in,
                              void* d_out, int out_size, void* d_ws, size_t ws_size,
                              hipStream_t stream) {
    const float* q    = (const float*)d_in[0];
    const float* k    = (const float*)d_in[1];
    const float* v    = (const float*)d_in[2];
    const int*   mask = (const int*)  d_in[3];
    float*       out  = (float*)d_out;

    dim3 grid(BB * HH * (LL / 128));   // 512 blocks: one per 128 q-rows of (b,h)
    dim3 block(256);                   // 8 wave32s
    hipLaunchKernelGGL(revattn_kernel, grid, block, 0, stream,
                       q, k, v, mask, out);
}
